// MultilabelHead_56633438765711
// MI455X (gfx1250) — matrix operand order repaired and checked by reading, not compiled, not run on hardware
//
#include <hip/hip_runtime.h>
#include <hip/hip_bf16.h>
#include <math.h>

typedef __attribute__((ext_vector_type(16))) __bf16 v16bf;
typedef __attribute__((ext_vector_type(8)))  float  v8f;

#define INPUT_DIM 1024
#define NUM_TARGETS 128
#define ROWS_PER_BLOCK 128   // 8 waves * 16 rows
#define KTILES (INPUT_DIM / 32)
#define CTILES (NUM_TARGETS / 16)

// ---------------------------------------------------------------------------
// Kernel 1: pack W [1024 x 128] f32 (row-major, W[k][t]) into bf16 B-fragments
// for v_wmma_f32_16x16x32_bf16.
// B fragment layout (16-bit B 32x16, wave32): lane holds col n = lane&15;
// element j holds K = (j&7) + 16*(j>>3) + 8*(lane>>4) (matches A fragment order).
// ws layout: frag[ct][kt][lane][j]  (j = 0..15 bf16), 8*32*32*16 bf16 = 256 KB.
// ---------------------------------------------------------------------------
__global__ __launch_bounds__(256) void pack_W_kernel(const float* __restrict__ W,
                                                     __bf16* __restrict__ ws) {
    int tid = blockIdx.x * blockDim.x + threadIdx.x;   // 0 .. 8191
    if (tid >= CTILES * KTILES * 32) return;
    int lane = tid & 31;
    int kt   = (tid >> 5) & (KTILES - 1);
    int ct   = tid >> 10;
    int n    = lane & 15;
    int hi   = lane >> 4;
    int t    = ct * 16 + n;
    int kbase = kt * 32 + hi * 8;
    __bf16* dst = ws + (size_t)tid * 16;
#pragma unroll
    for (int j = 0; j < 16; ++j) {
        dst[j] = (__bf16)W[(size_t)(kbase + (j & 7) + 16 * (j >> 3)) * NUM_TARGETS + t];
    }
}

// ---------------------------------------------------------------------------
// Kernel 2: streaming GEMM + bias + sigmoid.
// Block = 256 threads = 8 wave32s; each wave computes 16 rows x 128 cols.
// A fragment (16-bit A 16x32): lane holds row m = lane&15;
//   element j maps to k = (j&7) + 16*(j>>3) + 8*(lane>>4)
//   -> two contiguous 8-float runs at offsets (8*hi) and (16 + 8*hi).
// ---------------------------------------------------------------------------
__global__ __launch_bounds__(256) void gemm_sigmoid_kernel(
    const float* __restrict__ x, const __bf16* __restrict__ wfrag,
    const float* __restrict__ bias, float* __restrict__ out) {
    const int lane = threadIdx.x & 31;
    const int wv   = threadIdx.x >> 5;
    const int hi   = lane >> 4;
    const int m    = lane & 15;
    const int rowBase = blockIdx.x * ROWS_PER_BLOCK + wv * 16;

    const float* xrow = x + (size_t)(rowBase + m) * INPUT_DIM + hi * 8;

    v8f acc[CTILES];
#pragma unroll
    for (int ct = 0; ct < CTILES; ++ct) acc[ct] = (v8f){};

    for (int kt = 0; kt < KTILES; ++kt) {
        const float* p = xrow + kt * 32;
        const float4 a0 = *(const float4*)(p + 0);   // k = kb + 8*hi + 0..3
        const float4 a1 = *(const float4*)(p + 4);   // k = kb + 8*hi + 4..7
        const float4 a2 = *(const float4*)(p + 16);  // k = kb + 16 + 8*hi + 0..3
        const float4 a3 = *(const float4*)(p + 20);  // k = kb + 16 + 8*hi + 4..7
        v16bf a;
        a[0]  = (__bf16)a0.x; a[1]  = (__bf16)a0.y; a[2]  = (__bf16)a0.z; a[3]  = (__bf16)a0.w;
        a[4]  = (__bf16)a1.x; a[5]  = (__bf16)a1.y; a[6]  = (__bf16)a1.z; a[7]  = (__bf16)a1.w;
        a[8]  = (__bf16)a2.x; a[9]  = (__bf16)a2.y; a[10] = (__bf16)a2.z; a[11] = (__bf16)a2.w;
        a[12] = (__bf16)a3.x; a[13] = (__bf16)a3.y; a[14] = (__bf16)a3.z; a[15] = (__bf16)a3.w;

#pragma unroll
        for (int ct = 0; ct < CTILES; ++ct) {
            const v16bf b = *(const v16bf*)(wfrag + ((size_t)(ct * KTILES + kt) * 32 + lane) * 16);
            acc[ct] = __builtin_amdgcn_wmma_f32_16x16x32_bf16(
                false, a, false, b, (short)0, acc[ct], false, false);
        }
    }

    // Epilogue: bias + sigmoid, C/D layout: VGPR r -> row (r + 8*hi), col = lane&15.
    const int t0 = lane & 15;
#pragma unroll
    for (int ct = 0; ct < CTILES; ++ct) {
        const float bv = bias[ct * 16 + t0];
#pragma unroll
        for (int r = 0; r < 8; ++r) {
            const float z = acc[ct][r] + bv;
            const float s = 1.0f / (1.0f + __expf(-z));
            out[(size_t)(rowBase + r + hi * 8) * NUM_TARGETS + ct * 16 + t0] = s;
        }
    }
}

extern "C" void kernel_launch(void* const* d_in, const int* in_sizes, int n_in,
                              void* d_out, int out_size, void* d_ws, size_t ws_size,
                              hipStream_t stream) {
    const float* x    = (const float*)d_in[0];
    const float* W    = (const float*)d_in[1];
    const float* bias = (const float*)d_in[2];
    float* out        = (float*)d_out;
    __bf16* wfrag     = (__bf16*)d_ws;   // 256 KB of packed bf16 B-fragments

    const int batch = in_sizes[0] / INPUT_DIM;          // 32768
    const int nblocks = batch / ROWS_PER_BLOCK;         // 256

    pack_W_kernel<<<(CTILES * KTILES * 32 + 255) / 256, 256, 0, stream>>>(W, wfrag);
    gemm_sigmoid_kernel<<<nblocks, 256, 0, stream>>>(x, wfrag, bias, out);
}
